// CausalAttn_47261820125771
// MI455X (gfx1250) — matrix operand order repaired and checked by reading, not compiled, not run on hardware
//
#include <hip/hip_runtime.h>
#include <stdint.h>

// Problem constants (from reference): B=4, S=2048, D_IN=D_OUT=1024
#define BB 4
#define SS 2048
#define DD 1024
#define MM (BB * SS) // 8192 flattened (b,s) rows
#define KC 512       // AV-GEMM K slab width (LDS ping-pong chunk)

typedef __attribute__((ext_vector_type(16))) __bf16 v16bf;
typedef __attribute__((ext_vector_type(8)))  __bf16 v8bf;
typedef __attribute__((ext_vector_type(8)))  float  v8f;
typedef __attribute__((ext_vector_type(4)))  unsigned short us4;

// ---------------------------------------------------------------- helpers --

__device__ __forceinline__ unsigned short f2bf(float f) {
  unsigned int u = __float_as_uint(f);
  u += 0x7FFFu + ((u >> 16) & 1u); // round-to-nearest-even
  return (unsigned short)(u >> 16);
}

// Async copy 16 bytes global -> LDS (CDNA5 GLOBAL_LOAD_ASYNC_TO_LDS_B128,
// tracked by ASYNCcnt).  lds_off is the wave-relative LDS byte offset
// (= low 32 bits of the generic pointer to a __shared__ object).
__device__ __forceinline__ void async_g2l_b128(unsigned int lds_off, const void* g) {
  asm volatile("global_load_async_to_lds_b128 %0, %1, off"
               :: "v"(lds_off), "v"((unsigned long long)(uintptr_t)g)
               : "memory");
}
__device__ __forceinline__ void wait_async0() {
  asm volatile("s_wait_asynccnt 0x0" ::: "memory");
}

// A-matrix fragment (16x32 bf16, MxK): lane&15 = M row; kh = (lane>>4)*8:
// VGPR0-3 hold K = kh..kh+7, VGPR4-7 hold K = 16+kh..16+kh+7.
// Works for global or LDS base pointers (LDS -> ds_load_b128).
__device__ __forceinline__ v16bf frag_a(const unsigned short* base,
                                        int row0, int ldk, int k0, int lane) {
  const int lr = lane & 15;
  const int kh = (lane >> 4) << 3;
  const unsigned short* p = base + (size_t)(row0 + lr) * ldk + (k0 + kh);
  union { v16bf v; v8bf h[2]; } f;
  f.h[0] = *(const v8bf*)(p);
  f.h[1] = *(const v8bf*)(p + 16);
  return f.v;
}

// B-matrix fragment (32x16 bf16, KxN): lane&15 = N column; kh = (lane>>4)*8:
// VGPR0-3 hold K = kh..kh+7, VGPR4-7 hold K = 16+kh..16+kh+7.
__device__ __forceinline__ v16bf frag_b(const unsigned short* __restrict__ base,
                                        int row0, int ldk, int k0, int lane) {
  const unsigned short* p =
      base + (size_t)(row0 + (lane & 15)) * ldk + (k0 + ((lane >> 4) << 3));
  union { v16bf v; v8bf h[2]; } f;
  f.h[0] = *(const v8bf*)(p);
  f.h[1] = *(const v8bf*)(p + 16);
  return f.v;
}

#define WMMA_BF16(A, Bm, C) \
  __builtin_amdgcn_wmma_f32_16x16x32_bf16(false, (A), false, (Bm), (short)0, (C), false, false)

// ---------------------------------------------------------------- kernels --

// fp32 -> bf16, 4 elements per thread
__global__ __launch_bounds__(256) void cvt_f32_to_bf16(
    const float* __restrict__ src, unsigned short* __restrict__ dst, int n4) {
  int i = blockIdx.x * blockDim.x + threadIdx.x;
  if (i >= n4) return;
  float4 f = ((const float4*)src)[i];
  us4 o;
  o[0] = f2bf(f.x); o[1] = f2bf(f.y); o[2] = f2bf(f.z); o[3] = f2bf(f.w);
  ((us4*)dst)[i] = o;
}

// C[m,n] = sum_k A[m,k] * W[n,k]   (A:[8192,1024] bf16, W:[1024,1024] bf16)
// 4 waves/block share one 32-row A tile (async-staged to LDS, 64 KB); each
// wave computes a 32x64 strip: 8 WMMAs per pipelined K-step, B fragments
// amortized over two A rows-of-tiles.  K-loop unrolled 2x so the register
// ping-pong resolves by renaming (no v_mov rotation).
// vTranspose: store result as Vt[b][n][s] instead of C[m][n].
__global__ __launch_bounds__(128) void qkv_gemm(
    const unsigned short* __restrict__ A, const unsigned short* __restrict__ W,
    unsigned short* __restrict__ Out, int vTranspose) {
  __shared__ unsigned short sA[32 * DD]; // 64 KB
  const int t = threadIdx.x;
  const int lane = t & 31;
  const int wid = blockIdx.x * 4 + (t >> 5); // 0..4095
  const int mt = wid >> 4;       // 256 m-tiles of 32 (shared by the 4 waves)
  const int nstrip = wid & 15;   // 16 n-strips of 64
  const int m0 = mt << 5, n0 = nstrip << 6;

  { // stage A tile: rows m0..m0+31 contiguous = 64 KB = 4096 x 16B
    const unsigned short* gA = A + (size_t)m0 * DD;
    for (int c = t; c < 4096; c += 128)
      async_g2l_b128((unsigned int)(uintptr_t)&sA[c * 8], gA + c * 8);
  }
  wait_async0();
  __syncthreads();

  v8f acc[2][4] = {};
  v16bf a[2], b[4];
#pragma unroll
  for (int i = 0; i < 2; ++i) a[i] = frag_a(sA, i << 4, DD, 0, lane);
#pragma unroll
  for (int j = 0; j < 4; ++j) b[j] = frag_b(W, n0 + (j << 4), DD, 0, lane);

#pragma unroll 2
  for (int k0 = 0; k0 < DD - 32; k0 += 32) {
    const int kn = k0 + 32;
    v16bf na[2], nb[4];
#pragma unroll
    for (int i = 0; i < 2; ++i) na[i] = frag_a(sA, i << 4, DD, kn, lane);
#pragma unroll
    for (int j = 0; j < 4; ++j) nb[j] = frag_b(W, n0 + (j << 4), DD, kn, lane);
#pragma unroll
    for (int i = 0; i < 2; ++i)
#pragma unroll
      for (int j = 0; j < 4; ++j)
        acc[i][j] = WMMA_BF16(a[i], b[j], acc[i][j]);
#pragma unroll
    for (int i = 0; i < 2; ++i) a[i] = na[i];
#pragma unroll
    for (int j = 0; j < 4; ++j) b[j] = nb[j];
  }
#pragma unroll
  for (int i = 0; i < 2; ++i)
#pragma unroll
    for (int j = 0; j < 4; ++j)
      acc[i][j] = WMMA_BF16(a[i], b[j], acc[i][j]);

  const int cr = (lane >> 4) << 3; // row offset 0 or 8
  const int cn = lane & 15;
#pragma unroll
  for (int i = 0; i < 2; ++i)
#pragma unroll
    for (int j = 0; j < 4; ++j) {
      const int n = n0 + (j << 4) + cn;
#pragma unroll
      for (int r = 0; r < 8; ++r) {
        const int m = m0 + (i << 4) + cr + r;
        const unsigned short h = f2bf(acc[i][j][r]);
        if (!vTranspose) {
          Out[(size_t)m * DD + n] = h;
        } else {
          const int bb = m >> 11, s = m & (SS - 1);
          Out[((size_t)bb * DD + n) * SS + s] = h;
        }
      }
    }
}

// Sc[b][i][j] = (Q[b][i].K[b][j])/32.  Each wave: 32 rows x 64 cols of the
// score matrix.  4 waves/block share the 32-row Q tile (LDS staged).
// Strips entirely above the diagonal are skipped; partial strips may write
// garbage at j>i which softmax never reads.
__global__ __launch_bounds__(128) void scores_gemm(
    const unsigned short* __restrict__ Q, const unsigned short* __restrict__ K,
    float* __restrict__ Sc) {
  __shared__ unsigned short sQ[32 * DD]; // 64 KB
  const int t = threadIdx.x;
  const int lane = t & 31;
  const int wid = blockIdx.x * 4 + (t >> 5); // 0..8191
  const int b = wid >> 11;       // 2048 waves per batch
  const int rem = wid & 2047;
  const int it = rem >> 5;       // 64 i-tiles of 32 rows (shared by block)
  const int jstrip = rem & 31;   // 32 j-strips of 64
  const unsigned short* Qb = Q + (size_t)b * SS * DD;
  const unsigned short* Kb = K + (size_t)b * SS * DD;

  { // stage Q tile (contiguous 64 KB)
    const unsigned short* gQ = Qb + ((size_t)it << 5) * DD;
    for (int c = t; c < 4096; c += 128)
      async_g2l_b128((unsigned int)(uintptr_t)&sQ[c * 8], gQ + c * 8);
  }
  wait_async0();
  __syncthreads();

  const int j0 = jstrip << 6;
  if (j0 > (it << 5) + 31) return; // wave-uniform: strip fully above diagonal

  v8f acc[2][4] = {};
  v16bf a[2], b4[4];
#pragma unroll
  for (int i = 0; i < 2; ++i) a[i] = frag_a(sQ, i << 4, DD, 0, lane);
#pragma unroll
  for (int j = 0; j < 4; ++j) b4[j] = frag_b(Kb, j0 + (j << 4), DD, 0, lane);

#pragma unroll 2
  for (int k0 = 0; k0 < DD - 32; k0 += 32) {
    const int kn = k0 + 32;
    v16bf na[2], nb[4];
#pragma unroll
    for (int i = 0; i < 2; ++i) na[i] = frag_a(sQ, i << 4, DD, kn, lane);
#pragma unroll
    for (int j = 0; j < 4; ++j) nb[j] = frag_b(Kb, j0 + (j << 4), DD, kn, lane);
#pragma unroll
    for (int i = 0; i < 2; ++i)
#pragma unroll
      for (int j = 0; j < 4; ++j)
        acc[i][j] = WMMA_BF16(a[i], b4[j], acc[i][j]);
#pragma unroll
    for (int i = 0; i < 2; ++i) a[i] = na[i];
#pragma unroll
    for (int j = 0; j < 4; ++j) b4[j] = nb[j];
  }
#pragma unroll
  for (int i = 0; i < 2; ++i)
#pragma unroll
    for (int j = 0; j < 4; ++j)
      acc[i][j] = WMMA_BF16(a[i], b4[j], acc[i][j]);

  float* out = Sc + (size_t)b * SS * SS;
  const int cr = (lane >> 4) << 3;
  const int cn = lane & 15;
  const float scale = 1.0f / 32.0f; // 1/sqrt(D_IN)
#pragma unroll
  for (int i = 0; i < 2; ++i)
#pragma unroll
    for (int j = 0; j < 4; ++j) {
      const int jj = j0 + (j << 4) + cn;
#pragma unroll
      for (int r = 0; r < 8; ++r)
        out[(size_t)((it << 5) + (i << 4) + cr + r) * SS + jj] = acc[i][j][r] * scale;
    }
}

// Row-wise causal softmax: one wave per query row; reads fp32 scores[0..s],
// writes bf16 probabilities, zero-fills j>s so AV GEMM needs no masking.
__global__ __launch_bounds__(128) void softmax_rows(
    const float* __restrict__ Sc, unsigned short* __restrict__ P) {
  const int lane = threadIdx.x & 31;
  const int wid = blockIdx.x * 4 + (threadIdx.x >> 5); // 0..8191
  const int b = wid >> 11, s = wid & (SS - 1);
  const float* row = Sc + (size_t)b * SS * SS + (size_t)s * SS;
  unsigned short* prow = P + (size_t)b * SS * SS + (size_t)s * SS;

  float mx = -3.4e38f;
  for (int j = lane; j <= s; j += 32) mx = fmaxf(mx, row[j]);
#pragma unroll
  for (int off = 16; off > 0; off >>= 1) mx = fmaxf(mx, __shfl_xor(mx, off, 32));

  float sum = 0.0f;
  for (int j = lane; j <= s; j += 32) sum += __expf(row[j] - mx);
#pragma unroll
  for (int off = 16; off > 0; off >>= 1) sum += __shfl_xor(sum, off, 32);

  const float inv = 1.0f / sum;
  for (int j = lane; j < SS; j += 32) {
    const float p = (j <= s) ? __expf(row[j] - mx) * inv : 0.0f;
    prow[j] = f2bf(p);
  }
}

// O[b][s][n] = sum_k P[b][s][k] * Vt[b][n][k]; K-loop causally bounded per
// 32-row M-tile (kend = (mt+1)*32).  The 32-row P tile is streamed through
// LDS in KC-wide slabs with ping-pong buffers: async-prefetch slab c+1 while
// WMMAs consume slab c (ASYNCcnt producer/consumer pattern).
__global__ __launch_bounds__(128) void av_gemm(
    const unsigned short* __restrict__ P, const unsigned short* __restrict__ Vt,
    float* __restrict__ O) {
  __shared__ unsigned short sP[2][32 * KC]; // 2 x 32 KB
  const int t = threadIdx.x;
  const int lane = t & 31;
  const int wid = blockIdx.x * 4 + (t >> 5); // 0..4095
  const int b = wid >> 10;      // 1024 waves per batch
  const int rem = wid & 1023;
  const int mt = rem >> 4;      // 64 m-tiles of 32 rows (shared by block)
  const int nstrip = rem & 15;  // 16 n-strips of 64
  const int m0 = mt << 5, n0 = nstrip << 6;
  const unsigned short* Pb = P + (size_t)b * SS * SS;
  const unsigned short* Vb = Vt + (size_t)b * DD * SS;
  const int kend = (mt + 1) << 5;            // causal bound, multiple of 32
  const int nchunks = (kend + KC - 1) / KC;

  { // stage slab 0
    const int cw = kend < KC ? kend : KC;
    const int cpr = cw >> 3; // 16B chunks per row
    for (int c = t; c < 32 * cpr; c += 128) {
      const int row = c / cpr, col = c - row * cpr;
      async_g2l_b128((unsigned int)(uintptr_t)&sP[0][row * KC + col * 8],
                     Pb + (size_t)(m0 + row) * SS + col * 8);
    }
  }
  wait_async0();
  __syncthreads();

  v8f acc[2][4] = {};
  for (int c = 0; c < nchunks; ++c) {
    const int kbase = c * KC;
    const int cw = (kend - kbase) < KC ? (kend - kbase) : KC;
    const unsigned short* buf = sP[c & 1];

    if (c + 1 < nchunks) { // async-prefetch next slab into the other buffer
      const int nwid = (kend - kbase - KC) < KC ? (kend - kbase - KC) : KC;
      const int cpr = nwid >> 3;
      unsigned short* dbuf = sP[(c + 1) & 1];
      for (int cc = t; cc < 32 * cpr; cc += 128) {
        const int row = cc / cpr, col = cc - row * cpr;
        async_g2l_b128((unsigned int)(uintptr_t)&dbuf[row * KC + col * 8],
                       Pb + (size_t)(m0 + row) * SS + (kbase + KC) + col * 8);
      }
    }

    v16bf a[2], b4[4];
#pragma unroll
    for (int i = 0; i < 2; ++i) a[i] = frag_a(buf, i << 4, KC, 0, lane);
#pragma unroll
    for (int j = 0; j < 4; ++j) b4[j] = frag_b(Vb, n0 + (j << 4), SS, kbase, lane);

#pragma unroll 2
    for (int k0 = 0; k0 < cw - 32; k0 += 32) {
      const int kn = k0 + 32;
      v16bf na[2], nb[4];
#pragma unroll
      for (int i = 0; i < 2; ++i) na[i] = frag_a(buf, i << 4, KC, kn, lane);
#pragma unroll
      for (int j = 0; j < 4; ++j) nb[j] = frag_b(Vb, n0 + (j << 4), SS, kbase + kn, lane);
#pragma unroll
      for (int i = 0; i < 2; ++i)
#pragma unroll
        for (int j = 0; j < 4; ++j)
          acc[i][j] = WMMA_BF16(a[i], b4[j], acc[i][j]);
#pragma unroll
      for (int i = 0; i < 2; ++i) a[i] = na[i];
#pragma unroll
      for (int j = 0; j < 4; ++j) b4[j] = nb[j];
    }
#pragma unroll
    for (int i = 0; i < 2; ++i)
#pragma unroll
      for (int j = 0; j < 4; ++j)
        acc[i][j] = WMMA_BF16(a[i], b4[j], acc[i][j]);

    wait_async0();   // my prefetch writes have landed
    __syncthreads(); // everyone done reading buf & writing other buffer
  }

  float* Ob = O + (size_t)b * SS * DD;
  const int cr = (lane >> 4) << 3;
  const int cn = lane & 15;
#pragma unroll
  for (int i = 0; i < 2; ++i)
#pragma unroll
    for (int j = 0; j < 4; ++j) {
      const int n = n0 + (j << 4) + cn;
#pragma unroll
      for (int r = 0; r < 8; ++r)
        Ob[(size_t)(m0 + (i << 4) + cr + r) * DD + n] = acc[i][j][r];
    }
}

// ----------------------------------------------------------------- launch --

extern "C" void kernel_launch(void* const* d_in, const int* in_sizes, int n_in,
                              void* d_out, int out_size, void* d_ws, size_t ws_size,
                              hipStream_t stream) {
  (void)in_sizes; (void)n_in; (void)out_size; (void)ws_size;
  const float* x  = (const float*)d_in[0];
  const float* Wk = (const float*)d_in[1];
  const float* Wq = (const float*)d_in[2];
  const float* Wv = (const float*)d_in[3];

  // Workspace layout (ushort units unless noted):
  unsigned short* ws  = (unsigned short*)d_ws;
  unsigned short* xb  = ws;                          // x   bf16 [8192,1024]
  unsigned short* wkb = xb  + (size_t)MM * DD;       // Wk  bf16 [1024,1024]
  unsigned short* wqb = wkb + (size_t)DD * DD;       // Wq  bf16
  unsigned short* wvb = wqb + (size_t)DD * DD;       // Wv  bf16
  unsigned short* qb  = wvb + (size_t)DD * DD;       // Q   bf16 [8192,1024]
  unsigned short* kb  = qb  + (size_t)MM * DD;       // K   bf16 [8192,1024]
  unsigned short* vtb = kb  + (size_t)MM * DD;       // V^T bf16 [4][1024][2048]
  float*          sc  = (float*)(vtb + (size_t)MM * DD); // scores fp32 [4][2048][2048]
  unsigned short* pb  = (unsigned short*)(sc + (size_t)BB * SS * SS); // P bf16

  // 1) convert inputs to bf16
  cvt_f32_to_bf16<<<(MM * DD / 4) / 256, 256, 0, stream>>>(x, xb, MM * DD / 4);
  cvt_f32_to_bf16<<<(DD * DD / 4) / 256, 256, 0, stream>>>(Wk, wkb, DD * DD / 4);
  cvt_f32_to_bf16<<<(DD * DD / 4) / 256, 256, 0, stream>>>(Wq, wqb, DD * DD / 4);
  cvt_f32_to_bf16<<<(DD * DD / 4) / 256, 256, 0, stream>>>(Wv, wvb, DD * DD / 4);

  // 2) QKV projections (V stored transposed for the AV GEMM)
  qkv_gemm<<<1024, 128, 0, stream>>>(xb, wqb, qb, 0);
  qkv_gemm<<<1024, 128, 0, stream>>>(xb, wkb, kb, 0);
  qkv_gemm<<<1024, 128, 0, stream>>>(xb, wvb, vtb, 1);

  // 3) causal score strips (Q K^T / 32), fp32 (L2-resident: 67 MB << 192 MB)
  scores_gemm<<<2048, 128, 0, stream>>>(qb, kb, sc);

  // 4) row softmax -> P (bf16, zero-filled past causal bound)
  softmax_rows<<<2048, 128, 0, stream>>>(sc, pb);

  // 5) O = P V (causally bounded, LDS ping-pong slabs), fp32 to d_out
  av_gemm<<<1024, 128, 0, stream>>>(pb, vtb, (float*)d_out);
}
